// DeepVCP_64931315581273
// MI455X (gfx1250) — matrix operand order repaired and checked by reading, not compiled, not run on hardware
//
#include <hip/hip_runtime.h>
#include <cstddef>

// ---------------- CDNA5 WMMA fragment helpers (wave32) ----------------
typedef __attribute__((ext_vector_type(16))) _Float16 v16h;
typedef __attribute__((ext_vector_type(8)))  _Float16 v8h;
typedef __attribute__((ext_vector_type(8)))  float    v8f;

union Frag16 { v16h v; v8h p[2]; _Float16 h[16]; };
union Frag8f { v8f  v; float f[8]; };

__device__ __forceinline__ int lane_id() { return (int)(threadIdx.x & 31u); }

// A fragment: 16x32 f16 tile, rows at `src` (row stride `ld` halves), col offset k0.
// Layout (ISA 7.12.2): M = lane%16; halves 0..7 -> K = kb+h, halves 8..15 -> K = kb+16+(h-8),
// kb = (lane>=16)?8:0.  Each lane = two contiguous 16-byte runs -> two b128 loads.
__device__ __forceinline__ v16h load_A_f16(const _Float16* src, int ld, int k0) {
  const int L  = lane_id();
  const int m  = L & 15;
  const int kb = (L >> 4) << 3;                 // 0 or 8
  const _Float16* p = src + m * ld + k0 + kb;   // 16B aligned by construction
  Frag16 a;
  a.p[0] = *(const v8h*)(p);
  a.p[1] = *(const v8h*)(p + 16);
  return a.v;
}

// B fragment: 32x16 chunk of an LDS-staged, N-major, pre-zero-padded f16 weight
// Wl[n][k] with row stride Kpad halves. Layout: N = lane%16; halves 0..7 -> K = kb+h, halves 8..15 -> K = kb+16+(h-8), kb = (lane>=16)?8:0.
// two runs of 8 halves contiguous in K -> two b128 LDS loads, branch-free.
__device__ __forceinline__ v16h load_B_lds(const _Float16* Wl, int Kpad, int k0, int n0) {
  const int L  = lane_id();
  const int n  = (L & 15) + n0;
  const int kb = (L >> 4) << 3;                 // 0 or 8
  const _Float16* p = Wl + n * Kpad + k0 + kb;
  Frag16 b;
  b.p[0] = *(const v8h*)(p);
  b.p[1] = *(const v8h*)(p + 16);
  return b.v;
}

// Cooperative staging: f32 row-major W (Krows x Nfull) -> f16 N-major dst[n*Kpad+k],
// zero-padded for k >= Krows (done once; removes per-WMMA bounds branches).
__device__ __forceinline__ void stage_W(_Float16* dst, const float* W, int Nfull,
                                        int Krows, int Kpad, int nthreads, int tid) {
  for (int e = tid; e < Nfull * Kpad; e += nthreads) {
    int n = e / Kpad, k = e - n * Kpad;
    dst[e] = (_Float16)((k < Krows) ? W[k * Nfull + n] : 0.0f);
  }
}

// C/D f32 16x16: M = reg + (lane>=16?8:0), N = lane%16.
__device__ __forceinline__ int c_row(int r) { return r + ((lane_id() >> 4) << 3); }
__device__ __forceinline__ int c_col()      { return lane_id() & 15; }

__device__ __forceinline__ v8f bias_acc(const float* bias, int n0) {
  float bv = bias[n0 + c_col()];
  Frag8f c;
#pragma unroll
  for (int r = 0; r < 8; ++r) c.f[r] = bv;
  return c.v;
}

__device__ __forceinline__ v8f wmma16(v16h a, v16h b, v8f c) {
  return __builtin_amdgcn_wmma_f32_16x16x32_f16(false, a, false, b, (short)0, c,
                                                false, false);
}

// ---------------- Problem constants ----------------
#define NB   2
#define NC   6
#define NPT  2048
#define NK   64
#define NS   32
#define FD   32
#define NCAND 216        // 6^3 grid offsets
#define NCAND_PAD 224    // 14 M-tiles of 16

// ================= Kernel 1: point-feature MLP (6 -> 64 -> 32) =================
// One wave (32 threads) per 16-point tile. grid = (N/16, B).
__global__ __launch_bounds__(32) void fe_kernel(
    const float* __restrict__ pts,   // (B,6,N)
    const float* __restrict__ W1, const float* __restrict__ b1,
    const float* __restrict__ W2, const float* __restrict__ b2,
    const float* __restrict__ Wwl, const float* __restrict__ bwl,
    float* __restrict__ feat,        // (B,N,32)
    float* __restrict__ wout)        // (B,N) or nullptr
{
  __shared__ __align__(16) _Float16 lds_in[16 * 32];
  __shared__ __align__(16) _Float16 lds_h [16 * 64];
  __shared__ __align__(16) _Float16 W1l[64 * 32];   // N-major, K padded 6->32
  __shared__ __align__(16) _Float16 W2l[32 * 64];   // N-major, K = 64
  __shared__ float lds_f[16 * 32];

  const int tile = blockIdx.x;
  const int b    = blockIdx.y;
  const int m0   = tile * 16;
  const int tid  = threadIdx.x;

  __builtin_prefetch(W1, 0, 0);
  __builtin_prefetch(W2, 0, 0);

  stage_W(W1l, W1, 64, NC, 32, 32, tid);
  stage_W(W2l, W2, 32, 64, 64, 32, tid);

  // Stage 16x32 f16 input tile (6 channels, K zero-padded to 32).
  for (int e = tid; e < 16 * 32; e += 32) {
    int m = e >> 5, k = e & 31;
    float v = (k < NC) ? pts[(size_t)b * NC * NPT + (size_t)k * NPT + (m0 + m)]
                       : 0.0f;
    lds_in[e] = (_Float16)v;
  }
  __syncthreads();

  v16h a = load_A_f16(lds_in, 32, 0);
  // Layer 1: (16x32) @ (32x64).
#pragma unroll
  for (int t = 0; t < 4; ++t) {
    v8f acc = bias_acc(b1, t * 16);
    acc = wmma16(a, load_B_lds(W1l, 32, 0, t * 16), acc);
    Frag8f cf; cf.v = acc;
    int n = c_col() + t * 16;
#pragma unroll
    for (int r = 0; r < 8; ++r) {
      float v = cf.f[r]; v = v > 0.f ? v : 0.f;
      lds_h[c_row(r) * 64 + n] = (_Float16)v;
    }
  }
  __syncthreads();

  // Layer 2: (16x64) @ (64x32).
#pragma unroll
  for (int t = 0; t < 2; ++t) {
    v8f acc = bias_acc(b2, t * 16);
#pragma unroll
    for (int kc = 0; kc < 2; ++kc) {
      v16h a2 = load_A_f16(lds_h, 64, kc * 32);
      acc = wmma16(a2, load_B_lds(W2l, 64, kc * 32, t * 16), acc);
    }
    Frag8f cf; cf.v = acc;
    int n = c_col() + t * 16;
#pragma unroll
    for (int r = 0; r < 8; ++r) {
      float v = cf.f[r]; v = v > 0.f ? v : 0.f;
      int m = c_row(r);
      feat[((size_t)b * NPT + (m0 + m)) * FD + n] = v;
      lds_f[m * FD + n] = v;
    }
  }
  __syncthreads();

  if (wout && tid < 16) {
    float s = bwl[0];
#pragma unroll
    for (int d = 0; d < FD; ++d) s += lds_f[tid * FD + d] * Wwl[d];
    wout[(size_t)b * NPT + m0 + tid] = s;
  }
}

// ================= Kernel 2: top-K keypoints (iterative argmax) =================
__global__ __launch_bounds__(256) void topk_kernel(
    const float* __restrict__ w,        // (B,N)
    const float* __restrict__ src_pts,  // (B,6,N)
    float* __restrict__ out_keypts,     // (B,64,6) -> d_out
    float* __restrict__ centers)        // (B,64,3)
{
  __shared__ float lv[NPT];
  __shared__ float rv[256];
  __shared__ int   ri[256];
  const int b = blockIdx.x, tid = threadIdx.x;
  for (int i = tid; i < NPT; i += 256) lv[i] = w[b * NPT + i];
  __syncthreads();

  for (int it = 0; it < NK; ++it) {
    float best = -__builtin_inff(); int bi = 0x7fffffff;
    for (int i = tid; i < NPT; i += 256) {
      float v = lv[i];
      if (v > best || (v == best && i < bi)) { best = v; bi = i; }
    }
    rv[tid] = best; ri[tid] = bi;
    __syncthreads();
    for (int s = 128; s > 0; s >>= 1) {
      if (tid < s) {
        float v2 = rv[tid + s]; int i2 = ri[tid + s];
        if (v2 > rv[tid] || (v2 == rv[tid] && i2 < ri[tid])) { rv[tid] = v2; ri[tid] = i2; }
      }
      __syncthreads();
    }
    int idx = ri[0];
    if (tid < NC) {
      float v = src_pts[(size_t)b * NC * NPT + (size_t)tid * NPT + idx];
      out_keypts[(b * NK + it) * NC + tid] = v;
      if (tid < 3) centers[(b * NK + it) * 3 + tid] = v;
    }
    if (tid == 0) lv[idx] = -__builtin_inff();
    __syncthreads();
  }
}

// ====== Kernel 3: NS-nearest grouping + src_cat assembly (f16, padded K=64) ======
__global__ __launch_bounds__(256) void group_kernel(
    const float* __restrict__ src_pts,   // (B,6,N)
    const float* __restrict__ src_feat,  // (B,N,32)
    const float* __restrict__ centers,   // (B,64,3)
    _Float16* __restrict__ src_cat)      // (B*64*32, 64)
{
  __shared__ float d2[NPT];
  __shared__ float rv[256]; __shared__ int ri[256];
  __shared__ int   sel[NS]; __shared__ float seld[NS];
  const int bk = blockIdx.x, b = bk >> 6, tid = threadIdx.x;
  const float cx = centers[bk * 3], cy = centers[bk * 3 + 1], cz = centers[bk * 3 + 2];
  const float* px = src_pts + (size_t)b * NC * NPT;

  for (int i = tid; i < NPT; i += 256) {
    float dx = px[i] - cx, dy = px[NPT + i] - cy, dz = px[2 * NPT + i] - cz;
    d2[i] = dx * dx + dy * dy + dz * dz;
  }
  __syncthreads();

  for (int it = 0; it < NS; ++it) {     // iterative argmin, ties -> lower index
    float best = __builtin_inff(); int bi = 0x7fffffff;
    for (int i = tid; i < NPT; i += 256) {
      float v = d2[i];
      if (v < best || (v == best && i < bi)) { best = v; bi = i; }
    }
    rv[tid] = best; ri[tid] = bi; __syncthreads();
    for (int s = 128; s > 0; s >>= 1) {
      if (tid < s) {
        float v2 = rv[tid + s]; int i2 = ri[tid + s];
        if (v2 < rv[tid] || (v2 == rv[tid] && i2 < ri[tid])) { rv[tid] = v2; ri[tid] = i2; }
      }
      __syncthreads();
    }
    if (tid == 0) { sel[it] = ri[0]; seld[it] = rv[0]; d2[ri[0]] = __builtin_inff(); }
    __syncthreads();
  }

  const float* feat = src_feat + (size_t)b * NPT * FD;
  for (int e = tid; e < NS * 64; e += 256) {
    int j = e >> 6, t = e & 63;
    int idx = (seld[j] <= 1.0f) ? sel[j] : sel[0];   // RADIUS^2 = 1.0
    float v;
    if (t < 3)       v = px[t * NPT + idx] - centers[bk * 3 + t];
    else if (t < 35) v = feat[(size_t)idx * FD + (t - 3)];
    else             v = 0.0f;
    src_cat[(size_t)(bk * NS + j) * 64 + t] = (_Float16)v;
  }
}

// ===== Kernel 4: transform + 216 candidates + NN in tgt (tgt cloud in LDS) =====
__global__ __launch_bounds__(256) void candnn_kernel(
    const float* __restrict__ tgt_pts,  // (B,6,N)
    const float* __restrict__ tgt_feat, // (B,N,32)
    const float* __restrict__ centers,  // (B,64,3)
    const float* __restrict__ R,        // (3,3)
    const float* __restrict__ tv,       // (1,3)
    float* __restrict__ cand_out,       // (B*64*216, 3)
    _Float16* __restrict__ tgt_cat)     // (B*64, 224, 64)
{
  __shared__ float xyz[NPT * 3];        // 24 KB: whole target cloud resident
  const int bk = blockIdx.x, b = bk >> 6, tid = threadIdx.x;
  const float* px = tgt_pts + (size_t)b * NC * NPT;
  for (int i = tid; i < NPT; i += 256) {
    xyz[i * 3 + 0] = px[i];
    xyz[i * 3 + 1] = px[NPT + i];
    xyz[i * 3 + 2] = px[2 * NPT + i];
  }
  __syncthreads();

  if (tid < NCAND) {
    float cx = centers[bk * 3], cy = centers[bk * 3 + 1], cz = centers[bk * 3 + 2];
    float qx = R[0] * cx + R[1] * cy + R[2] * cz + tv[0];
    float qy = R[3] * cx + R[4] * cy + R[5] * cz + tv[1];
    float qz = R[6] * cx + R[7] * cy + R[8] * cz + tv[2];
    int gi = tid / 36, gj = (tid / 6) % 6, gk = tid % 6;
    qx += gi * 0.4f - 1.0f; qy += gj * 0.4f - 1.0f; qz += gk * 0.4f - 1.0f;

    float best = __builtin_inff(); int bi = 0;
    for (int n = 0; n < NPT; ++n) {     // first-occurrence min == jnp.argmin
      float dx = xyz[n * 3] - qx, dy = xyz[n * 3 + 1] - qy, dz = xyz[n * 3 + 2] - qz;
      float d = dx * dx + dy * dy + dz * dz;
      if (d < best) { best = d; bi = n; }
    }

    size_t row = (size_t)bk * NCAND + tid;
    cand_out[row * 3 + 0] = qx; cand_out[row * 3 + 1] = qy; cand_out[row * 3 + 2] = qz;

    const float* feat = tgt_feat + ((size_t)b * NPT + bi) * FD;
    _Float16* dst = tgt_cat + ((size_t)bk * NCAND_PAD + tid) * 64;
    dst[0] = (_Float16)(xyz[bi * 3 + 0] - qx);
    dst[1] = (_Float16)(xyz[bi * 3 + 1] - qy);
    dst[2] = (_Float16)(xyz[bi * 3 + 2] - qz);
#pragma unroll
    for (int d = 0; d < FD; ++d) dst[3 + d] = (_Float16)feat[d];
#pragma unroll
    for (int p = 35; p < 64; ++p) dst[p] = (_Float16)0.0f;
  }
}

// ====== Kernel 5: dfe on src_cat (35->64->32 WMMA) + max over NS rows ======
// One wave per (b,k): 32 rows = 2 M-tiles.
__global__ __launch_bounds__(32) void srcdfe_kernel(
    const _Float16* __restrict__ src_cat, // (B*64*32, 64)
    const float* __restrict__ W3, const float* __restrict__ b3,
    const float* __restrict__ W4, const float* __restrict__ b4,
    float* __restrict__ src_dfe)          // (B*64, 32)
{
  __shared__ __align__(16) _Float16 W3l[64 * 64];   // N-major, K padded 35->64
  __shared__ __align__(16) _Float16 W4l[32 * 64];   // N-major, K = 64
  __shared__ __align__(16) _Float16 lds_h[16 * 64];
  __shared__ float dout[NS * FD];
  const int bk = blockIdx.x, tid = threadIdx.x;
  const _Float16* cat = src_cat + (size_t)bk * NS * 64;

  __builtin_prefetch(W3, 0, 0);
  __builtin_prefetch(W4, 0, 0);
  stage_W(W3l, W3, 64, 35, 64, 32, tid);
  stage_W(W4l, W4, 32, 64, 64, 32, tid);
  __syncthreads();

  for (int mt = 0; mt < 2; ++mt) {
    const _Float16* arows = cat + (size_t)mt * 16 * 64;
#pragma unroll
    for (int t = 0; t < 4; ++t) {        // layer 1: K padded 35->64
      v8f acc = bias_acc(b3, t * 16);
#pragma unroll
      for (int kc = 0; kc < 2; ++kc) {
        v16h a = load_A_f16(arows, 64, kc * 32);
        acc = wmma16(a, load_B_lds(W3l, 64, kc * 32, t * 16), acc);
      }
      Frag8f cf; cf.v = acc;
      int n = c_col() + t * 16;
#pragma unroll
      for (int r = 0; r < 8; ++r) {
        float v = cf.f[r]; v = v > 0.f ? v : 0.f;
        lds_h[c_row(r) * 64 + n] = (_Float16)v;
      }
    }
    __syncthreads();
#pragma unroll
    for (int t = 0; t < 2; ++t) {        // layer 2: 64 -> 32
      v8f acc = bias_acc(b4, t * 16);
#pragma unroll
      for (int kc = 0; kc < 2; ++kc) {
        v16h a = load_A_f16(lds_h, 64, kc * 32);
        acc = wmma16(a, load_B_lds(W4l, 64, kc * 32, t * 16), acc);
      }
      Frag8f cf; cf.v = acc;
      int n = c_col() + t * 16;
#pragma unroll
      for (int r = 0; r < 8; ++r) {
        float v = cf.f[r]; v = v > 0.f ? v : 0.f;
        dout[(mt * 16 + c_row(r)) * FD + n] = v;
      }
    }
    __syncthreads();
  }

  int d = tid;                           // max over the 32 group rows
  float m = dout[d];
  for (int r = 1; r < NS; ++r) m = fmaxf(m, dout[r * FD + d]);
  src_dfe[(size_t)bk * FD + d] = m;
}

// === Kernel 6: dfe on tgt_cat (WMMA) + sim + softmax + vcp, fused per (b,k) ===
// 256 threads (8 waves); 14 M-tiles of 16 rows; wave-uniform WMMA guards.
__global__ __launch_bounds__(256) void tgtdfe_kernel(
    const _Float16* __restrict__ tgt_cat, // (B*64, 224, 64)
    const float* __restrict__ W3, const float* __restrict__ b3,
    const float* __restrict__ W4, const float* __restrict__ b4,
    const float* __restrict__ src_dfe,    // (B*64, 32)
    const float* __restrict__ cand,       // (B*64, 216, 3)
    float* __restrict__ vcp_out)          // (B*64, 3) -> d_out + 768
{
  __shared__ __align__(16) _Float16 W3l[64 * 64];     // 8 KB
  __shared__ __align__(16) _Float16 W4l[32 * 64];     // 4 KB
  __shared__ __align__(16) _Float16 lds_h[8][16 * 64];// 16 KB, per-wave hidden
  __shared__ float tdfe[NCAND_PAD * FD];              // 28 KB
  __shared__ float sdfe[FD];
  __shared__ float red[256];
  const int bk = blockIdx.x, tid = threadIdx.x, wave = tid >> 5;

  __builtin_prefetch(W3, 0, 0);
  __builtin_prefetch(W4, 0, 0);
  stage_W(W3l, W3, 64, 35, 64, 256, tid);
  stage_W(W4l, W4, 32, 64, 64, 256, tid);
  if (tid < FD) sdfe[tid] = src_dfe[(size_t)bk * FD + tid];
  __syncthreads();

  const _Float16* cat = tgt_cat + (size_t)bk * NCAND_PAD * 64;

  for (int it = 0; it < 2; ++it) {       // uniform trip count; barriers uniform
    const int mt = it * 8 + wave;
    const bool active = (mt < 14);       // wave-uniform -> EXEC stays all-ones
    if (active) {
      const _Float16* arows = cat + (size_t)mt * 16 * 64;
#pragma unroll
      for (int t = 0; t < 4; ++t) {
        v8f acc = bias_acc(b3, t * 16);
#pragma unroll
        for (int kc = 0; kc < 2; ++kc) {
          v16h a = load_A_f16(arows, 64, kc * 32);
          acc = wmma16(a, load_B_lds(W3l, 64, kc * 32, t * 16), acc);
        }
        Frag8f cf; cf.v = acc;
        int n = c_col() + t * 16;
#pragma unroll
        for (int r = 0; r < 8; ++r) {
          float v = cf.f[r]; v = v > 0.f ? v : 0.f;
          lds_h[wave][c_row(r) * 64 + n] = (_Float16)v;
        }
      }
    }
    __syncthreads();
    if (active) {
#pragma unroll
      for (int t = 0; t < 2; ++t) {
        v8f acc = bias_acc(b4, t * 16);
#pragma unroll
        for (int kc = 0; kc < 2; ++kc) {
          v16h a = load_A_f16(lds_h[wave], 64, kc * 32);
          acc = wmma16(a, load_B_lds(W4l, 64, kc * 32, t * 16), acc);
        }
        Frag8f cf; cf.v = acc;
        int n = c_col() + t * 16;
#pragma unroll
        for (int r = 0; r < 8; ++r) {
          float v = cf.f[r]; v = v > 0.f ? v : 0.f;
          tdfe[(mt * 16 + c_row(r)) * FD + n] = v;
        }
      }
    }
    __syncthreads();
  }

  // sim = tgt_dfe . src_dfe ; softmax over 216 ; vcp = sum attn * cand
  float simv = -__builtin_inff();
  if (tid < NCAND) {
    float s = 0.f;
#pragma unroll
    for (int d = 0; d < FD; ++d) s += tdfe[tid * FD + d] * sdfe[d];
    simv = s;
  }
  red[tid] = simv; __syncthreads();
  for (int s = 128; s > 0; s >>= 1) {
    if (tid < s) red[tid] = fmaxf(red[tid], red[tid + s]);
    __syncthreads();
  }
  float mx = red[0]; __syncthreads();

  float e = (tid < NCAND) ? __expf(simv - mx) : 0.f;
  red[tid] = e; __syncthreads();
  for (int s = 128; s > 0; s >>= 1) {
    if (tid < s) red[tid] += red[tid + s];
    __syncthreads();
  }
  float esum = red[0]; __syncthreads();

  const float* cd = cand + (size_t)bk * NCAND * 3;
  for (int i = 0; i < 3; ++i) {
    float p = (tid < NCAND) ? e * cd[tid * 3 + i] : 0.f;
    red[tid] = p; __syncthreads();
    for (int s = 128; s > 0; s >>= 1) {
      if (tid < s) red[tid] += red[tid + s];
      __syncthreads();
    }
    if (tid == 0) vcp_out[bk * 3 + i] = red[0] / esum;
    __syncthreads();
  }
}

// ================================ launch ================================
extern "C" void kernel_launch(void* const* d_in, const int* in_sizes, int n_in,
                              void* d_out, int out_size, void* d_ws, size_t ws_size,
                              hipStream_t stream) {
  (void)in_sizes; (void)n_in; (void)out_size; (void)ws_size;
  const float* src_pts = (const float*)d_in[0];
  const float* tgt_pts = (const float*)d_in[1];
  const float* R       = (const float*)d_in[2];
  const float* tvec    = (const float*)d_in[3];
  const float* W1      = (const float*)d_in[4];
  const float* b1      = (const float*)d_in[5];
  const float* W2      = (const float*)d_in[6];
  const float* b2      = (const float*)d_in[7];
  const float* Wwl     = (const float*)d_in[8];
  const float* bwl     = (const float*)d_in[9];
  const float* W3      = (const float*)d_in[10];
  const float* b3      = (const float*)d_in[11];
  const float* W4      = (const float*)d_in[12];
  const float* b4      = (const float*)d_in[13];
  float* out = (float*)d_out;

  char* ws = (char*)d_ws;
  size_t off = 0;
  auto alloc = [&](size_t bytes) -> char* {
    char* p = ws + off;
    off = (off + bytes + 255) & ~(size_t)255;
    return p;
  };
  float*    src_feat = (float*)   alloc((size_t)NB * NPT * FD * 4);
  float*    tgt_feat = (float*)   alloc((size_t)NB * NPT * FD * 4);
  float*    wbuf     = (float*)   alloc((size_t)NB * NPT * 4);
  float*    centers  = (float*)   alloc((size_t)NB * NK * 3 * 4);
  _Float16* src_cat  = (_Float16*)alloc((size_t)NB * NK * NS * 64 * 2);
  _Float16* tgt_cat  = (_Float16*)alloc((size_t)NB * NK * NCAND_PAD * 64 * 2);
  float*    cand     = (float*)   alloc((size_t)NB * NK * NCAND * 3 * 4);
  float*    src_dfe  = (float*)   alloc((size_t)NB * NK * FD * 4);

  dim3 feg(NPT / 16, NB);
  fe_kernel<<<feg, 32, 0, stream>>>(src_pts, W1, b1, W2, b2, Wwl, bwl,
                                    src_feat, wbuf);
  fe_kernel<<<feg, 32, 0, stream>>>(tgt_pts, W1, b1, W2, b2, Wwl, bwl,
                                    tgt_feat, nullptr);
  topk_kernel<<<NB, 256, 0, stream>>>(wbuf, src_pts, out, centers);
  group_kernel<<<NB * NK, 256, 0, stream>>>(src_pts, src_feat, centers, src_cat);
  candnn_kernel<<<NB * NK, 256, 0, stream>>>(tgt_pts, tgt_feat, centers, R, tvec,
                                             cand, tgt_cat);
  srcdfe_kernel<<<NB * NK, 32, 0, stream>>>(src_cat, W3, b3, W4, b4, src_dfe);
  tgtdfe_kernel<<<NB * NK, 256, 0, stream>>>(tgt_cat, W3, b3, W4, b4, src_dfe,
                                             cand, out + NB * NK * NC);
}
